// Many2Many_28278064677260
// MI455X (gfx1250) — compile-verified, matrix operand order checked by reading, not run on hardware
//
#include <hip/hip_runtime.h>

// ---------------------------------------------------------------------------
// 2-layer LSTM + fused FC for MI455X (gfx1250), bf16 WMMA, fp32 accumulate.
// B=128, S=1024, F=8, H=512, O=8.
// ---------------------------------------------------------------------------

typedef __attribute__((ext_vector_type(16))) __bf16        v16bf;
typedef __attribute__((ext_vector_type(8)))  float         v8f;
typedef __attribute__((ext_vector_type(4)))  unsigned int  u32x4;

union FragU {
  v16bf          b;
  u32x4          q[2];
  unsigned short h[16];
};

__device__ __forceinline__ unsigned short f2bf(float f) {
  unsigned u = __float_as_uint(f);
  unsigned r = u + 0x7FFFu + ((u >> 16) & 1u);   // round-to-nearest-even
  return (unsigned short)(r >> 16);
}

// A (16x32, M=lane&15) and B (32x16, N=lane&15) bf16 fragment loader.
// Lanes 0-15 hold K = {k0..k0+7} and {k0+16..k0+23}; lanes 16-31 are shifted
// by kh*8 which the caller folds into k0.  Two 16B loads per fragment.
__device__ __forceinline__ v16bf load_frag(const unsigned short* rowp, int k0) {
  FragU f;
  f.q[0] = *reinterpret_cast<const u32x4*>(rowp + k0);
  f.q[1] = *reinterpret_cast<const u32x4*>(rowp + k0 + 16);
  return f.b;
}

__device__ __forceinline__ float sigf(float x) { return 1.0f / (1.0f + __expf(-x)); }

#define WMMA_BF16(A, Bf, C) \
  __builtin_amdgcn_wmma_f32_16x16x32_bf16(false, (A), false, (Bf), (short)0, (C), false, false)

// ---------------------------------------------------------------------------
// FC tile: pred[t] = h1[t] @ Wfc^T + bfc   (128x512 @ 512x8, W padded to 16)
// One workgroup, 8 waves, each wave one M-tile, 16 K-steps of WMMA.
// h1 lives in columns 0..511 of the [128][1024] bf16 concat buffer.
// ---------------------------------------------------------------------------
__device__ __forceinline__ void fc_step(const unsigned short* __restrict__ h1cat,
                                        const unsigned short* __restrict__ Wfc16,
                                        const float* __restrict__ bfc,
                                        float* __restrict__ out, int t) {
  const int wv = threadIdx.x >> 5;
  const int lane = threadIdx.x & 31;
  const int ln = lane & 15, kh = lane >> 4;
  const unsigned short* Brow = Wfc16 + (size_t)ln * 512;
  const unsigned short* Arow = h1cat + (size_t)(wv * 16 + ln) * 1024;
  v8f acc = {0.f, 0.f, 0.f, 0.f, 0.f, 0.f, 0.f, 0.f};
#pragma unroll
  for (int kt = 0; kt < 16; ++kt) {
    const int k0 = kt * 32 + kh * 8;
    v16bf bf = load_frag(Brow, k0);
    v16bf af = load_frag(Arow, k0);
    acc = WMMA_BF16(af, bf, acc);
  }
  if (ln < 8) {
    const float bb = bfc[ln];
#pragma unroll
    for (int v = 0; v < 8; ++v) {
      const int row = wv * 16 + kh * 8 + v;            // batch index
      out[((size_t)row * 1024 + t) * 8 + ln] = acc[v] + bb;
    }
  }
}

// ---------------------------------------------------------------------------
// One LSTM layer step.  gates = Acat @ Wcat^T + b, then pointwise.
//   IS_L0: Acat = [h0_{t-1} (512) | x_t (8) | pad (24)]  (K=544, x handled as
//          a tail K-tile built in registers), writes h0_t to h0nxt and to the
//          upper half of the layer-1 concat buffer.  Block 64 runs the fused
//          FC for step t-1.
//   !IS_L0: Acat = [h1_{t-1} | h0_t] = hc1cur (K=1024); writes h1_t into the
//          lower half of hc1nxt.
// Grid: 64 (+1 for L0) blocks x 256 threads.  Block = 64 batch rows x 16
// hidden units -> 4 gate-tiles x 4 M-tiles of 16x16, 2 tiles per wave.
// ---------------------------------------------------------------------------
template <bool IS_L0>
__global__ void lstm_step_kernel(const float* __restrict__ x,            // [B,S,F], L0 only
                                 const unsigned short* __restrict__ W,   // [2048][LDW] bf16
                                 const float* __restrict__ bsum,         // [2048]
                                 const unsigned short* __restrict__ Ain, // L0: h0 cur; L1: hc1 cur
                                 unsigned short* __restrict__ h0nxt,     // L0 only
                                 unsigned short* __restrict__ hc1w,      // L0: hc1 cur (cols 512+); L1: hc1 nxt (cols 0..511)
                                 float* __restrict__ c,                  // [128][512] fp32, in place
                                 const unsigned short* __restrict__ Wfc16,
                                 const float* __restrict__ bfc,
                                 float* __restrict__ out,
                                 int t) {
  if (IS_L0 && blockIdx.x == 64) {                     // fused FC for step t-1
    if (t > 0) fc_step(hc1w, Wfc16, bfc, out, t - 1);
    return;
  }

  constexpr int LDW = IS_L0 ? 544 : 1024;
  constexpr int LDA = IS_L0 ? 512 : 1024;
  constexpr int KT  = IS_L0 ? 16  : 32;                // full 32-wide K-tiles

  __shared__ float ldsG[64][64];                       // gates tile, 16 KB

  const int blk  = blockIdx.x;
  const int mh   = blk & 1;                            // which 64-row half of B
  const int hid0 = (blk >> 1) * 16;                    // hidden-unit base
  const int wv   = threadIdx.x >> 5;
  const int lane = threadIdx.x & 31;
  const int ln   = lane & 15, kh = lane >> 4;
  const int nt   = wv >> 1;                            // gate index 0..3 (i,f,g,o)
  const int mts  = (wv & 1) * 2;                       // first of this wave's 2 M-tiles

  const unsigned short* Brow = W + (size_t)(nt * 512 + hid0 + ln) * LDW;
  const int r0 = mh * 64 + mts * 16 + ln;
  const int r1 = r0 + 16;
  const unsigned short* Arow0 = Ain + (size_t)r0 * LDA;
  const unsigned short* Arow1 = Ain + (size_t)r1 * LDA;

  v8f acc0 = {0.f, 0.f, 0.f, 0.f, 0.f, 0.f, 0.f, 0.f};
  v8f acc1 = {0.f, 0.f, 0.f, 0.f, 0.f, 0.f, 0.f, 0.f};

#pragma unroll 4
  for (int kt = 0; kt < KT; ++kt) {
    const int k0 = kt * 32 + kh * 8;
    if (kt + 1 < KT) __builtin_prefetch(Brow + k0 + 32, 0, 1);   // global_prefetch
    v16bf bf = load_frag(Brow, k0);
    v16bf a0 = load_frag(Arow0, k0);
    v16bf a1 = load_frag(Arow1, k0);
    acc0 = WMMA_BF16(a0, bf, acc0);
    acc1 = WMMA_BF16(a1, bf, acc1);
  }

  if (IS_L0) {
    // Tail K-tile (k = 512..543): A = [x_t (8 cols) | zeros]; W rows carry W_ih0.
    v16bf bf = load_frag(Brow, 512 + kh * 8);
    FragU fa0, fa1;
    fa0.q[0] = (u32x4){0, 0, 0, 0}; fa0.q[1] = (u32x4){0, 0, 0, 0};
    fa1.q[0] = (u32x4){0, 0, 0, 0}; fa1.q[1] = (u32x4){0, 0, 0, 0};
    if (kh == 0) {                                    // lanes holding K=512..519
      union { unsigned short h[8]; u32x4 q; } p0, p1;
      const float* xr0 = x + ((size_t)r0 * 1024 + t) * 8;
      const float* xr1 = x + ((size_t)r1 * 1024 + t) * 8;
#pragma unroll
      for (int j = 0; j < 8; ++j) { p0.h[j] = f2bf(xr0[j]); p1.h[j] = f2bf(xr1[j]); }
      fa0.q[0] = p0.q;
      fa1.q[0] = p1.q;
    }
    acc0 = WMMA_BF16(fa0.b, bf, acc0);
    acc1 = WMMA_BF16(fa1.b, bf, acc1);
  }

  // Stage gate accumulators to LDS (C layout: VGPR v -> M = kh*8+v, N = ln).
#pragma unroll
  for (int v = 0; v < 8; ++v) {
    ldsG[mts * 16 + kh * 8 + v][nt * 16 + ln]       = acc0[v];
    ldsG[(mts + 1) * 16 + kh * 8 + v][nt * 16 + ln] = acc1[v];
  }
  __syncthreads();

  // Pointwise LSTM cell for this block's 64 rows x 16 hidden units.
#pragma unroll
  for (int s = 0; s < 4; ++s) {
    const int idx = threadIdx.x + s * 256;            // 0..1023
    const int r = idx >> 4;                           // local row 0..63
    const int j = idx & 15;                           // local hidden 0..15
    const int R = mh * 64 + r;                        // batch row
    const int hj = hid0 + j;                          // hidden unit
    const float iv = sigf (ldsG[r][j]      + bsum[hj]);
    const float fv = sigf (ldsG[r][16 + j] + bsum[512  + hj]);
    const float gv = tanhf(ldsG[r][32 + j] + bsum[1024 + hj]);
    const float ov = sigf (ldsG[r][48 + j] + bsum[1536 + hj]);
    const size_t ci = (size_t)R * 512 + hj;
    const float cv = fv * c[ci] + iv * gv;
    c[ci] = cv;
    const unsigned short hb = f2bf(ov * tanhf(cv));
    if (IS_L0) {
      h0nxt[ci] = hb;                                 // layer-0 state (next step)
      hc1w[(size_t)R * 1024 + 512 + hj] = hb;         // feed layer 1 this step
    } else {
      hc1w[(size_t)R * 1024 + hj] = hb;               // layer-1 state (next step)
    }
  }
}

// Final FC for the last timestep.
__global__ void fc_final_kernel(const unsigned short* __restrict__ h1cat,
                                const unsigned short* __restrict__ Wfc16,
                                const float* __restrict__ bfc,
                                float* __restrict__ out) {
  fc_step(h1cat, Wfc16, bfc, out, 1023);
}

// ---------------------------------------------------------------------------
// Prep: bf16-convert + concatenate weights, sum biases, zero all state.
// Wcat0[n][0..511]=W_hh0[n], [512..519]=W_ih0[n], [520..543]=0    (2048x544)
// Wcat1[n][0..511]=W_hh1[n], [512..1023]=W_ih1[n]                 (2048x1024)
// Wfc16[16][512] rows 8..15 zero.  b0/b1 = b_ih + b_hh (fp32).
// ---------------------------------------------------------------------------
__global__ void prep_kernel(const float* __restrict__ Wih0, const float* __restrict__ Whh0,
                            const float* __restrict__ bih0, const float* __restrict__ bhh0,
                            const float* __restrict__ Wih1, const float* __restrict__ Whh1,
                            const float* __restrict__ bih1, const float* __restrict__ bhh1,
                            const float* __restrict__ Wfcin,
                            unsigned short* __restrict__ Wcat0, unsigned short* __restrict__ Wcat1,
                            unsigned short* __restrict__ Wfc16,
                            float* __restrict__ b0, float* __restrict__ b1,
                            unsigned int* __restrict__ zbase) {
  const long T0 = 2048L * 544, T1 = 2048L * 1024, T2 = 16L * 512;
  const long T3 = 2048, T4 = 2048, TZ = 327680;       // state zero-fill (dwords)
  const long total = T0 + T1 + T2 + T3 + T4 + TZ;
  for (long i = blockIdx.x * (long)blockDim.x + threadIdx.x; i < total;
       i += (long)gridDim.x * blockDim.x) {
    long j = i;
    if (j < T0) {
      const int n = (int)(j / 544), k = (int)(j % 544);
      const float v = (k < 512) ? Whh0[(long)n * 512 + k]
                    : (k < 520) ? Wih0[(long)n * 8 + (k - 512)] : 0.0f;
      Wcat0[j] = f2bf(v);
    } else if ((j -= T0) < T1) {
      const int n = (int)(j / 1024), k = (int)(j % 1024);
      const float v = (k < 512) ? Whh1[(long)n * 512 + k] : Wih1[(long)n * 512 + (k - 512)];
      Wcat1[j] = f2bf(v);
    } else if ((j -= T1) < T2) {
      const int n = (int)(j / 512), k = (int)(j % 512);
      Wfc16[j] = f2bf(n < 8 ? Wfcin[(long)n * 512 + k] : 0.0f);
    } else if ((j -= T2) < T3) {
      b0[j] = bih0[j] + bhh0[j];
    } else if ((j -= T3) < T4) {
      b1[j] = bih1[j] + bhh1[j];
    } else {
      j -= T4;
      zbase[j] = 0u;
    }
  }
}

// ---------------------------------------------------------------------------
// Host side: carve workspace, prep once, then S steps of (layer0, layer1).
// ---------------------------------------------------------------------------
extern "C" void kernel_launch(void* const* d_in, const int* in_sizes, int n_in,
                              void* d_out, int out_size, void* d_ws, size_t ws_size,
                              hipStream_t stream) {
  (void)in_sizes; (void)n_in; (void)out_size; (void)ws_size;
  const float* x    = (const float*)d_in[0];
  const float* Wih0 = (const float*)d_in[1];
  const float* Whh0 = (const float*)d_in[2];
  const float* bih0 = (const float*)d_in[3];
  const float* bhh0 = (const float*)d_in[4];
  const float* Wih1 = (const float*)d_in[5];
  const float* Whh1 = (const float*)d_in[6];
  const float* bih1 = (const float*)d_in[7];
  const float* bhh1 = (const float*)d_in[8];
  const float* Wfc  = (const float*)d_in[9];
  const float* bfc  = (const float*)d_in[10];
  float* out = (float*)d_out;

  unsigned char* w = (unsigned char*)d_ws;
  size_t off = 0;
  auto carve = [&](size_t bytes) -> void* {
    void* p = w + off;
    off = (off + bytes + 255) & ~(size_t)255;
    return p;
  };
  unsigned short* Wcat0 = (unsigned short*)carve(2048UL * 544 * 2);
  unsigned short* Wcat1 = (unsigned short*)carve(2048UL * 1024 * 2);
  unsigned short* Wfc16 = (unsigned short*)carve(16UL * 512 * 2);
  float*          b0    = (float*)carve(2048UL * 4);
  float*          b1    = (float*)carve(2048UL * 4);
  // State block (contiguous; all sizes multiples of 256B -> zbase covers it):
  unsigned short* h0buf = (unsigned short*)carve(2UL * 128 * 512 * 2);   // 256 KB
  unsigned short* hc1   = (unsigned short*)carve(2UL * 128 * 1024 * 2);  // 512 KB
  float*          c0    = (float*)carve(128UL * 512 * 4);                // 256 KB
  float*          c1    = (float*)carve(128UL * 512 * 4);                // 256 KB
  unsigned int*   zbase = (unsigned int*)h0buf;       // 327680 dwords of state

  prep_kernel<<<4096, 256, 0, stream>>>(Wih0, Whh0, bih0, bhh0, Wih1, Whh1, bih1, bhh1,
                                        Wfc, Wcat0, Wcat1, Wfc16, b0, b1, zbase);

  for (int t = 0; t < 1024; ++t) {
    const int cur = t & 1, nxt = cur ^ 1;
    const unsigned short* h0c = h0buf + (size_t)cur * 128 * 512;
    unsigned short*       h0n = h0buf + (size_t)nxt * 128 * 512;
    unsigned short*       hc  = hc1 + (size_t)cur * 128 * 1024;  // [h1_{t-1} | h0_t]
    unsigned short*       hn  = hc1 + (size_t)nxt * 128 * 1024;
    lstm_step_kernel<true><<<65, 256, 0, stream>>>(x, Wcat0, b0, h0c, h0n, hc, c0,
                                                   Wfc16, bfc, out, t);
    lstm_step_kernel<false><<<64, 256, 0, stream>>>(nullptr, Wcat1, b1, hc, nullptr, hn,
                                                    c1, nullptr, nullptr, nullptr, t);
  }
  // h1_{1023} lands in hc1 buffer 0 (1024 is even).
  fc_final_kernel<<<1, 256, 0, stream>>>(hc1, Wfc16, bfc, out);
}
